// GraphCrossAttn_60550448939723
// MI455X (gfx1250) — matrix operand order repaired and checked by reading, not compiled, not run on hardware
//
#include <hip/hip_runtime.h>

typedef unsigned short u16;
typedef __attribute__((ext_vector_type(16))) __bf16 v16bf;
typedef __attribute__((ext_vector_type(8)))  float  v8f;

constexpr int N_NODES = 20000;
constexpr int N_EDGES = 320000;
constexpr int RNA_F = 2000, PROT_F = 100;
constexpr int HID_D = 128, NHEAD = 4, HD_D = 512;
constexpr int EMB_D = 256;

#if __has_builtin(__builtin_amdgcn_global_load_async_to_lds_b128) && \
    __has_builtin(__builtin_amdgcn_s_wait_asynccnt)
#define USE_ASYNC_LDS 1
#else
#define USE_ASYNC_LDS 0
#endif

// ---------------------------------------------------------------- helpers
__device__ __forceinline__ u16 f2bf(float f) {
  unsigned u = __float_as_uint(f);
  u += 0x7FFFu + ((u >> 16) & 1u);            // round-to-nearest-even
  return (u16)(u >> 16);
}

__device__ __forceinline__ void atomicMaxFloat(float* addr, float val) {
  if (val >= 0.0f) atomicMax((int*)addr, __float_as_int(val));
  else             atomicMin((unsigned int*)addr, __float_as_uint(val));
}

union Frag16 { uint4 u[2]; v16bf v; };

#if USE_ASYNC_LDS
typedef int nat_int4 __attribute__((vector_size(16)));            // native v4i
typedef __attribute__((address_space(1))) nat_int4 g_nat4;        // global (AS1)
typedef __attribute__((address_space(3))) nat_int4 l_nat4;        // LDS (AS3)
__device__ __forceinline__ void async_b128(const u16* g, u16* l) {
  __builtin_amdgcn_global_load_async_to_lds_b128(
      (g_nat4*)g, (l_nat4*)l, 0, 0);
}
#endif

// ---------------------------------------------------------------- bf16 WMMA GEMM
// C[M,ldc] (+bias, +relu, +bf16 mirror) = A[M,Kp](bf16) * Bt[Npad,Kp](bf16)^T
// Block: 256 thr = 8 waves; tile 128x128, BK=32; wave tile 32x64 (2x4 frags).
// Double-buffered LDS (one barrier per K-step); tile copies go through the
// gfx1250 async-to-LDS engine when the toolchain exposes it.
template<bool HAS_BIAS, bool RELU, bool BF16_OUT>
__global__ __launch_bounds__(256) void gemm_bf16(
    const u16* __restrict__ A, const u16* __restrict__ Bt,
    float* __restrict__ C, u16* __restrict__ Cb,
    const float* __restrict__ bias,
    int M, int Nc, int Kp, int ldc)
{
  __shared__ __align__(16) u16 As[2][128 * 32];
  __shared__ __align__(16) u16 Bs[2][128 * 32];

  const int tid   = threadIdx.x;
  const int lane  = tid & 31;
  const int wave  = tid >> 5;
  const int wm    = wave >> 1;          // 0..3  (M direction)
  const int wn    = wave & 1;           // 0..1  (N direction)
  const int m0    = blockIdx.x * 128;
  const int n0    = blockIdx.y * 128;
  const int mrow  = lane & 15;
  const int khalf = lane >> 4;

  // global tile load mapping: thread -> (row lr, 16-elem half lh)
  const int lr = tid >> 1, lh = tid & 1;
  int arow = m0 + lr; if (arow > M - 1) arow = M - 1;        // clamp M edge
  const u16* gA = A  + (size_t)arow * Kp + lh * 16;
  const u16* gB = Bt + (size_t)(n0 + lr) * Kp + lh * 16;     // Bt rows padded
  const int soff = lr * 32 + lh * 16;

#if USE_ASYNC_LDS
  // prologue: tile 0 -> buffer 0 via async DMA
  async_b128(gA,     &As[0][soff]);
  async_b128(gA + 8, &As[0][soff + 8]);
  async_b128(gB,     &Bs[0][soff]);
  async_b128(gB + 8, &Bs[0][soff + 8]);
#else
  uint4 ra0 = *(const uint4*)gA,       ra1 = *(const uint4*)(gA + 8);
  uint4 rb0 = *(const uint4*)gB,       rb1 = *(const uint4*)(gB + 8);
#endif

  v8f acc[2][4] = {};

  for (int k0 = 0; k0 < Kp; k0 += 32) {
    const int p = (k0 >> 5) & 1;
#if USE_ASYNC_LDS
    __builtin_amdgcn_s_wait_asynccnt(0);   // buffer p fully written by DMA
    __syncthreads();
    if (k0 + 32 < Kp) {                    // kick off next tile into buffer 1-p
      async_b128(gA + k0 + 32, &As[1 - p][soff]);
      async_b128(gA + k0 + 40, &As[1 - p][soff + 8]);
      async_b128(gB + k0 + 32, &Bs[1 - p][soff]);
      async_b128(gB + k0 + 40, &Bs[1 - p][soff + 8]);
    }
#else
    // commit staged registers into buffer p, prefetch next tile into registers
    *(uint4*)&As[p][soff]     = ra0;
    *(uint4*)&As[p][soff + 8] = ra1;
    *(uint4*)&Bs[p][soff]     = rb0;
    *(uint4*)&Bs[p][soff + 8] = rb1;
    if (k0 + 32 < Kp) {
      ra0 = *(const uint4*)(gA + k0 + 32); ra1 = *(const uint4*)(gA + k0 + 40);
      rb0 = *(const uint4*)(gB + k0 + 32); rb1 = *(const uint4*)(gB + k0 + 40);
    }
    __syncthreads();
#endif

    Frag16 a[2], b[4];
#pragma unroll
    for (int mi = 0; mi < 2; ++mi) {
      // A frag: lanes0-15 K{0..7,16..23}, lanes16-31 K{8..15,24..31}
      const u16* s = &As[p][(wm * 32 + mi * 16 + mrow) * 32 + khalf * 8];
      a[mi].u[0] = *(const uint4*)s;
      a[mi].u[1] = *(const uint4*)(s + 16);
    }
#pragma unroll
    for (int ni = 0; ni < 4; ++ni) {
      // B frag: lanes0-15 K{0..7,16..23}, lanes16-31 K{8..15,24..31} (col = lane&15)
      const u16* s = &Bs[p][(wn * 64 + ni * 16 + mrow) * 32 + khalf * 8];
      b[ni].u[0] = *(const uint4*)s;
      b[ni].u[1] = *(const uint4*)(s + 16);
    }
#pragma unroll
    for (int mi = 0; mi < 2; ++mi)
#pragma unroll
      for (int ni = 0; ni < 4; ++ni)
        acc[mi][ni] = __builtin_amdgcn_wmma_f32_16x16x32_bf16(
            false, a[mi].v, false, b[ni].v, (short)0, acc[mi][ni], false, false);
  }

#pragma unroll
  for (int mi = 0; mi < 2; ++mi)
#pragma unroll
    for (int ni = 0; ni < 4; ++ni) {
      int ccol = n0 + wn * 64 + ni * 16 + mrow;
      if (ccol >= Nc) continue;
      float bv = HAS_BIAS ? bias[ccol] : 0.0f;
#pragma unroll
      for (int r = 0; r < 8; ++r) {
        int crow = m0 + wm * 32 + mi * 16 + khalf * 8 + r;
        if (crow < M) {
          float v = acc[mi][ni][r] + bv;
          if (RELU) v = fmaxf(v, 0.0f);
          C[(size_t)crow * ldc + ccol] = v;
          if (BF16_OUT) Cb[(size_t)crow * Nc + ccol] = f2bf(v);
        }
      }
    }
}

// ---------------------------------------------------------------- conversions
// W [K,N] f32 row-major -> Wt [Np,Kp] bf16 (transposed, zero-padded)
__global__ void conv_w_t(const float* __restrict__ W, u16* __restrict__ Wt,
                         int K, int Nn, int Kp, int Np) {
  int t = blockIdx.x * blockDim.x + threadIdx.x;
  if (t >= Np * Kp) return;
  int n = t / Kp, k = t % Kp;
  Wt[t] = (k < K && n < Nn) ? f2bf(W[(size_t)k * Nn + n]) : (u16)0;
}

// x [N,2100] -> x_rna_b [N,2016] bf16 (pad), x_prot_b [N,128] bf16 (pad)
__global__ void split_x(const float* __restrict__ x, u16* __restrict__ xr,
                        u16* __restrict__ xp, int total) {
  int t = blockIdx.x * blockDim.x + threadIdx.x;
  if (t >= total) return;
  int n = t / 2144, c = t % 2144;
  if (c < 2016)
    xr[(size_t)n * 2016 + c] = (c < RNA_F) ? f2bf(x[(size_t)n * 2100 + c]) : (u16)0;
  else {
    int cc = c - 2016;
    xp[(size_t)n * 128 + cc] = (cc < PROT_F) ? f2bf(x[(size_t)n * 2100 + RNA_F + cc]) : (u16)0;
  }
}

__global__ void f32_to_bf16_k(const float* __restrict__ in, u16* __restrict__ out, int n) {
  int t = blockIdx.x * blockDim.x + threadIdx.x;
  if (t < n) out[t] = f2bf(in[t]);
}

__global__ void fill_f32(float* __restrict__ p, float v, int n) {
  int t = blockIdx.x * blockDim.x + threadIdx.x;
  if (t < n) p[t] = v;
}

// ---------------------------------------------------------------- GAT sparse part
// per node: a_src[n,h]=<xs[n,h,:],a_s[h,:]>, a_dst likewise (8 waves: 4 src + 4 dst)
__global__ __launch_bounds__(256) void attn_coef(
    const float* __restrict__ xs, const float* __restrict__ xd,
    const float* __restrict__ a_s, const float* __restrict__ a_d,
    float* __restrict__ o_src, float* __restrict__ o_dst) {
  int n = blockIdx.x;
  int wave = threadIdx.x >> 5, lane = threadIdx.x & 31;
  int head = wave & 3;
  const float* X  = (wave < 4) ? xs : xd;
  const float* Av = (wave < 4) ? a_s : a_d;
  const float* row = X + (size_t)n * HD_D + head * HID_D;
  const float* av  = Av + head * HID_D;
  float s = 0.0f;
  for (int j = lane; j < HID_D; j += 32) s += row[j] * av[j];
#pragma unroll
  for (int o = 16; o; o >>= 1) s += __shfl_xor(s, o, 32);
  if (lane == 0) ((wave < 4) ? o_src : o_dst)[n * NHEAD + head] = s;
}

// pass1: e = leaky_relu(a_src[src]+a_dst[dst]); segment-max via atomics
__global__ void edge_pass1(const int* __restrict__ src, const int* __restrict__ dst,
                           const float* __restrict__ a_src, const float* __restrict__ a_dst,
                           float* __restrict__ ebuf, float* __restrict__ emax) {
  int t = blockIdx.x * blockDim.x + threadIdx.x;
  if (t >= N_EDGES * NHEAD) return;
  int e = t >> 2, hd = t & 3;
  float s = a_src[src[e] * NHEAD + hd] + a_dst[dst[e] * NHEAD + hd];
  s = (s > 0.0f) ? s : 0.2f * s;
  ebuf[t] = s;
  atomicMaxFloat(&emax[dst[e] * NHEAD + hd], s);
}

// pass2 (one wave per edge): w=exp(e-max); denom += w; msg[dst] += xs[src]*w
__global__ __launch_bounds__(256) void edge_pass2(
    const int* __restrict__ src, const int* __restrict__ dst,
    const float* __restrict__ xs, const float* __restrict__ ebuf,
    const float* __restrict__ emax, float* __restrict__ den,
    float* __restrict__ msg) {
  int gw = (blockIdx.x * 256 + threadIdx.x) >> 5;
  int lane = threadIdx.x & 31;
  if (gw >= N_EDGES) return;
  int s = src[gw], d = dst[gw];
  float w = 0.0f;
  if (lane < NHEAD) {
    w = __expf(ebuf[gw * NHEAD + lane] - emax[d * NHEAD + lane]);
    atomicAdd(&den[d * NHEAD + lane], w);
  }
  float wh = __shfl(w, lane >> 3, 32);     // head = (lane*16)>>7 = lane>>3
  const float* xr = xs  + (size_t)s * HD_D + lane * 16;
  float*       mr = msg + (size_t)d * HD_D + lane * 16;
#pragma unroll
  for (int j = 0; j < 16; ++j) atomicAdd(&mr[j], xr[j] * wh);
}

// h = relu(msg/denom + g_b + lin + l_b); write f32 + bf16
__global__ void finalize_node(const float* __restrict__ msg, const float* __restrict__ den,
                              const float* __restrict__ lin, const float* __restrict__ gb,
                              const float* __restrict__ lb, float* __restrict__ h,
                              u16* __restrict__ hb, int total) {
  int t = blockIdx.x * blockDim.x + threadIdx.x;
  if (t >= total) return;
  int c = t & (HD_D - 1), n = t >> 9;
  float v = msg[t] / (den[n * NHEAD + (c >> 7)] + 1e-16f) + gb[c] + lin[t] + lb[c];
  v = fmaxf(v, 0.0f);
  h[t] = v;
  hb[t] = f2bf(v);
}

// ---------------------------------------------------------------- host
extern "C" void kernel_launch(void* const* d_in, const int* in_sizes, int n_in,
                              void* d_out, int out_size, void* d_ws, size_t ws_size,
                              hipStream_t stream) {
  (void)in_sizes; (void)n_in; (void)out_size; (void)ws_size;

  const float* x      = (const float*)d_in[0];
  const int*   ei     = (const int*)d_in[1];
  const int*   src    = ei;
  const int*   dst    = ei + N_EDGES;
  const float* W_rna  = (const float*)d_in[2];
  const float* b_rna  = (const float*)d_in[3];
  const float* W_prot = (const float*)d_in[4];
  const float* b_prot = (const float*)d_in[5];
  const float* g1_ws  = (const float*)d_in[6];
  const float* g1_wd  = (const float*)d_in[7];
  const float* g1_as  = (const float*)d_in[8];
  const float* g1_ad  = (const float*)d_in[9];
  const float* g1_b   = (const float*)d_in[10];
  const float* l1_w   = (const float*)d_in[11];
  const float* l1_b   = (const float*)d_in[12];
  const float* g2_ws  = (const float*)d_in[13];
  const float* g2_wd  = (const float*)d_in[14];
  const float* g2_as  = (const float*)d_in[15];
  const float* g2_ad  = (const float*)d_in[16];
  const float* g2_b   = (const float*)d_in[17];
  const float* l2_w   = (const float*)d_in[18];
  const float* l2_b   = (const float*)d_in[19];
  const float* agg_w  = (const float*)d_in[20];
  const float* agg_b  = (const float*)d_in[21];
  const float* dr_w   = (const float*)d_in[22];
  const float* dr_b   = (const float*)d_in[23];
  const float* dp_w   = (const float*)d_in[24];
  const float* dp_b   = (const float*)d_in[25];
  const float* rr_w   = (const float*)d_in[26];
  const float* rr_b   = (const float*)d_in[27];
  const float* rp_w   = (const float*)d_in[28];
  const float* rp_b   = (const float*)d_in[29];

  // ---- workspace bump allocator
  char* wsb = (char*)d_ws;
  size_t off = 0;
  auto alloc = [&](size_t bytes) -> char* {
    char* p = wsb + off;
    off = (off + bytes + 255) & ~(size_t)255;
    return p;
  };

  u16* wt_rna  = (u16*)alloc((size_t)256 * 2016 * 2);
  u16* wt_prot = (u16*)alloc((size_t)256 * 128 * 2);
  u16* wt_g1s  = (u16*)alloc((size_t)512 * 512 * 2);
  u16* wt_g1d  = (u16*)alloc((size_t)512 * 512 * 2);
  u16* wt_l1   = (u16*)alloc((size_t)512 * 512 * 2);
  u16* wt_g2s  = (u16*)alloc((size_t)512 * 512 * 2);
  u16* wt_g2d  = (u16*)alloc((size_t)512 * 512 * 2);
  u16* wt_l2   = (u16*)alloc((size_t)512 * 512 * 2);
  u16* wt_agg  = (u16*)alloc((size_t)128 * 512 * 2);
  u16* wt_dr   = (u16*)alloc((size_t)256 * 128 * 2);
  u16* wt_dp   = (u16*)alloc((size_t)256 * 128 * 2);
  u16* wt_rr   = (u16*)alloc((size_t)2048 * 256 * 2);
  u16* wt_rp   = (u16*)alloc((size_t)128 * 256 * 2);

  // big region: padded bf16 inputs; later reused for xs/xd, then decode bufs
  char* big0 = alloc((size_t)N_NODES * 2016 * 2 + (size_t)N_NODES * 128 * 2); // 85.76 MB
  u16*   x_rna_b  = (u16*)big0;
  u16*   x_prot_b = (u16*)(big0 + (size_t)N_NODES * 2016 * 2);
  float* xs       = (float*)big0;                                   // 41.0 MB
  float* xd       = (float*)(big0 + (size_t)N_NODES * HD_D * 4);    // 41.0 MB (ends 82 MB)
  u16*   rna_d_b  = (u16*)big0;                                     // after GAT layers
  u16*   prot_d_b = (u16*)(big0 + (size_t)N_NODES * EMB_D * 2);
  u16*   emb_b    = (u16*)(big0 + (size_t)N_NODES * HD_D * 4);      // alias dead xd

  float* h    = (float*)alloc((size_t)N_NODES * HD_D * 4);
  u16*   hb   = (u16*)  alloc((size_t)N_NODES * HD_D * 2);
  float* lin  = (float*)alloc((size_t)N_NODES * HD_D * 4);
  float* msg  = (float*)alloc((size_t)N_NODES * HD_D * 4);
  float* asrc = (float*)alloc((size_t)N_NODES * NHEAD * 4);
  float* adst = (float*)alloc((size_t)N_NODES * NHEAD * 4);
  float* ebuf = (float*)alloc((size_t)N_EDGES * NHEAD * 4);
  float* emax = (float*)alloc((size_t)N_NODES * NHEAD * 4);
  float* den  = (float*)alloc((size_t)N_NODES * NHEAD * 4);
  float* rna_d_f  = lin;   // dead after layer-2 finalize
  float* prot_d_f = msg;

  auto cdiv = [](int a, int b) { return (a + b - 1) / b; };
  const int MB = cdiv(N_NODES, 128);   // 157
  dim3 gM1(MB, 1), gM2(MB, 2), gM4(MB, 4), gM16(MB, 16);

  // ---- weight conversions (f32 -> transposed padded bf16)
  conv_w_t<<<cdiv(256 * 2016, 256), 256, 0, stream>>>(W_rna,  wt_rna,  2000, 256, 2016, 256);
  conv_w_t<<<cdiv(256 * 128, 256),  256, 0, stream>>>(W_prot, wt_prot, 100,  256, 128,  256);
  conv_w_t<<<cdiv(512 * 512, 256),  256, 0, stream>>>(g1_ws,  wt_g1s,  512,  512, 512,  512);
  conv_w_t<<<cdiv(512 * 512, 256),  256, 0, stream>>>(g1_wd,  wt_g1d,  512,  512, 512,  512);
  conv_w_t<<<cdiv(512 * 512, 256),  256, 0, stream>>>(l1_w,   wt_l1,   512,  512, 512,  512);
  conv_w_t<<<cdiv(512 * 512, 256),  256, 0, stream>>>(g2_ws,  wt_g2s,  512,  512, 512,  512);
  conv_w_t<<<cdiv(512 * 512, 256),  256, 0, stream>>>(g2_wd,  wt_g2d,  512,  512, 512,  512);
  conv_w_t<<<cdiv(512 * 512, 256),  256, 0, stream>>>(l2_w,   wt_l2,   512,  512, 512,  512);
  conv_w_t<<<cdiv(128 * 512, 256),  256, 0, stream>>>(agg_w,  wt_agg,  512,  128, 512,  128);
  conv_w_t<<<cdiv(256 * 128, 256),  256, 0, stream>>>(dr_w,   wt_dr,   128,  256, 128,  256);
  conv_w_t<<<cdiv(256 * 128, 256),  256, 0, stream>>>(dp_w,   wt_dp,   128,  256, 128,  256);
  conv_w_t<<<cdiv(2048 * 256, 256), 256, 0, stream>>>(rr_w,   wt_rr,   256, 2000, 256, 2048);
  conv_w_t<<<cdiv(128 * 256, 256),  256, 0, stream>>>(rp_w,   wt_rp,   256,  100, 256,  128);

  split_x<<<cdiv(N_NODES * 2144, 256), 256, 0, stream>>>(x, x_rna_b, x_prot_b, N_NODES * 2144);

  // ---- input embeddings: h = [x_rna@W_rna+b, x_prot@W_prot+b]
  gemm_bf16<true, false, false><<<gM2, 256, 0, stream>>>(x_rna_b,  wt_rna,  h,       nullptr, b_rna,  N_NODES, 256, 2016, 512);
  gemm_bf16<true, false, false><<<gM2, 256, 0, stream>>>(x_prot_b, wt_prot, h + 256, nullptr, b_prot, N_NODES, 256, 128,  512);
  f32_to_bf16_k<<<cdiv(N_NODES * HD_D, 256), 256, 0, stream>>>(h, hb, N_NODES * HD_D);

  // ---- GAT layer (xs/xd/lin GEMMs + segment softmax + scatter + residual relu)
  auto gat = [&](const u16* wts, const u16* wtd, const u16* wtl,
                 const float* as_, const float* ad_, const float* gb_, const float* lb_) {
    gemm_bf16<false, false, false><<<gM4, 256, 0, stream>>>(hb, wts, xs,  nullptr, nullptr, N_NODES, 512, 512, 512);
    gemm_bf16<false, false, false><<<gM4, 256, 0, stream>>>(hb, wtd, xd,  nullptr, nullptr, N_NODES, 512, 512, 512);
    gemm_bf16<false, false, false><<<gM4, 256, 0, stream>>>(hb, wtl, lin, nullptr, nullptr, N_NODES, 512, 512, 512);
    attn_coef<<<N_NODES, 256, 0, stream>>>(xs, xd, as_, ad_, asrc, adst);
    fill_f32<<<cdiv(N_NODES * NHEAD, 256), 256, 0, stream>>>(emax, -__builtin_huge_valf(), N_NODES * NHEAD);
    fill_f32<<<cdiv(N_NODES * NHEAD, 256), 256, 0, stream>>>(den, 0.0f, N_NODES * NHEAD);
    fill_f32<<<cdiv(N_NODES * HD_D, 256), 256, 0, stream>>>(msg, 0.0f, N_NODES * HD_D);
    edge_pass1<<<cdiv(N_EDGES * NHEAD, 256), 256, 0, stream>>>(src, dst, asrc, adst, ebuf, emax);
    edge_pass2<<<N_EDGES / 8, 256, 0, stream>>>(src, dst, xs, ebuf, emax, den, msg);
    finalize_node<<<cdiv(N_NODES * HD_D, 256), 256, 0, stream>>>(msg, den, lin, gb_, lb_, h, hb, N_NODES * HD_D);
  };
  gat(wt_g1s, wt_g1d, wt_l1, g1_as, g1_ad, g1_b, l1_b);
  gat(wt_g2s, wt_g2d, wt_l2, g2_as, g2_ad, g2_b, l2_b);

  // ---- head: embedding + decoders + reconstruction
  float* out      = (float*)d_out;
  float* out_rna  = out;
  float* out_prot = out + (size_t)N_NODES * RNA_F;
  float* out_emb  = out + (size_t)N_NODES * (RNA_F + PROT_F);

  gemm_bf16<true, true,  true ><<<gM1,  256, 0, stream>>>(hb,       wt_agg, out_emb,  emb_b,    agg_b, N_NODES, 128,  512, 128);
  gemm_bf16<true, false, true ><<<gM2,  256, 0, stream>>>(emb_b,    wt_dr,  rna_d_f,  rna_d_b,  dr_b,  N_NODES, 256,  128, 256);
  gemm_bf16<true, false, true ><<<gM2,  256, 0, stream>>>(emb_b,    wt_dp,  prot_d_f, prot_d_b, dp_b,  N_NODES, 256,  128, 256);
  gemm_bf16<true, false, false><<<gM16, 256, 0, stream>>>(rna_d_b,  wt_rr,  out_rna,  nullptr,  rr_b,  N_NODES, 2000, 256, 2000);
  gemm_bf16<true, false, false><<<gM1,  256, 0, stream>>>(prot_d_b, wt_rp,  out_prot, nullptr,  rp_b,  N_NODES, 100,  256, 100);
}
